// CS_26499948216876
// MI455X (gfx1250) — matrix operand order repaired and checked by reading, not compiled, not run on hardware
//
#include <hip/hip_runtime.h>
#include <hip/hip_bf16.h>
#include <math.h>

#define B_  64
#define C_  2048
#define P_  196
#define PP_ 224          // P padded to multiple of 32 for the GEMM N dimension
#define HS_ 1024
#define K_  512
#define EPSV 1e-12f

typedef __attribute__((ext_vector_type(16))) __bf16          v16bf;
typedef __attribute__((ext_vector_type(16))) unsigned short  v16u;
typedef __attribute__((ext_vector_type(8)))  unsigned short  v8u;
typedef __attribute__((ext_vector_type(8)))  float           v8f;

__device__ __forceinline__ unsigned short f2bf(float f) {
    unsigned int u = __float_as_uint(f);
    unsigned int r = u + 0x7FFFu + ((u >> 16) & 1u);   // round-to-nearest-even
    return (unsigned short)(r >> 16);
}

__device__ __forceinline__ float wsum(float v) {
#pragma unroll
    for (int o = 16; o > 0; o >>= 1) v += __shfl_xor(v, o, 32);
    return v;
}

// ---- 1. per-(b,c) spatial L2 stats -> ic[b,c] = mean(x)/max(||x||,eps) ----
__global__ void k_ic(const float* __restrict__ img, float* __restrict__ ic) {
    int gw = blockIdx.x * 8 + (threadIdx.x >> 5);        // one wave per (b,c) row
    int l  = threadIdx.x & 31;
    size_t base = (size_t)gw * P_;
    float s = 0.f, q = 0.f;
    for (int i = l; i < P_; i += 32) { float v = img[base + i]; s += v; q += v * v; }
    s = wsum(s); q = wsum(q);
    if (l == 0) {
        float rn = 1.f / fmaxf(sqrtf(q), EPSV);
        ic[gw] = s * (1.f / (float)P_) * rn;
    }
}

// ---- 2. ws (f32) -> bf16 copy (L2-resident A matrix for WMMA) ----
__global__ void k_cvtws(const float* __restrict__ ws, unsigned short* __restrict__ wsb) {
    int i = blockIdx.x * 256 + threadIdx.x;
    wsb[i] = f2bf(ws[i]);
}

// ---- 3. hc = whc@h, hsv = whs@h  (one wave per output element) ----
__global__ void k_hproj(const float* __restrict__ whc, const float* __restrict__ whs,
                        const float* __restrict__ h, float* __restrict__ hc,
                        float* __restrict__ hsv) {
    int gw = blockIdx.x * 8 + (threadIdx.x >> 5);
    int l  = threadIdx.x & 31;
    int sel = gw >> 15;
    int rem = gw & 32767;
    int b = rem >> 9, k = rem & 511;
    const float* wr = (sel ? whs : whc) + (size_t)k * HS_;
    const float* hr = h + (size_t)b * HS_;
    float s = 0.f;
    for (int i = l; i < HS_; i += 32) s += wr[i] * hr[i];
    s = wsum(s);
    if (l == 0) (sel ? hsv : hc)[b * K_ + k] = s;
}

// ---- 4. sc[b,c] = sum_k wci[k]*tanh(wc[k]*ic + bc[k] + hc[b,k]) + bci ----
__global__ void k_sc(const float* __restrict__ ic, const float* __restrict__ hc,
                     const float* __restrict__ wc, const float* __restrict__ bc,
                     const float* __restrict__ wci, const float* __restrict__ bci,
                     float* __restrict__ sc) {
    int gw = blockIdx.x * 8 + (threadIdx.x >> 5);        // (b,c)
    int l  = threadIdx.x & 31;
    int b  = gw >> 11;
    float icv = ic[gw];
    const float* hcb = hc + b * K_;
    float s = 0.f;
    for (int k = l; k < K_; k += 32)
        s += wci[k] * tanhf(wc[k] * icv + bc[k] + hcb[k]);
    s = wsum(s);
    if (l == 0) sc[gw] = s + bci[0];
}

// ---- 5. softmax over C -> ac ----
__global__ void k_smC(const float* __restrict__ sc, float* __restrict__ ac) {
    __shared__ float red[256];
    int b = blockIdx.x, t = threadIdx.x;
    const float* row = sc + (size_t)b * C_;
    float v[8];
    float mx = -1e30f;
#pragma unroll
    for (int i = 0; i < 8; i++) { v[i] = row[t + 256 * i]; mx = fmaxf(mx, v[i]); }
    red[t] = mx; __syncthreads();
    for (int s = 128; s > 0; s >>= 1) { if (t < s) red[t] = fmaxf(red[t], red[t + s]); __syncthreads(); }
    mx = red[0]; __syncthreads();
    float sum = 0.f;
#pragma unroll
    for (int i = 0; i < 8; i++) { v[i] = expf(v[i] - mx); sum += v[i]; }
    red[t] = sum; __syncthreads();
    for (int s = 128; s > 0; s >>= 1) { if (t < s) red[t] += red[t + s]; __syncthreads(); }
    float rs = 1.f / red[0];
    float* orow = ac + (size_t)b * C_;
#pragma unroll
    for (int i = 0; i < 8; i++) orow[t + 256 * i] = v[i] * rs;
}

// ---- 6. xw = x*ac staged TRANSPOSED+PADDED as bf16 xwt[b][p][c];
//         plus deterministic per-chunk sum-of-squares partials ----
__global__ void k_xw(const float* __restrict__ img, const float* __restrict__ ac,
                     unsigned short* __restrict__ xwt, float* __restrict__ sqp) {
    __shared__ unsigned short ldsT[16 * 200];    // [c_local][p], stride 200
    int chunk = blockIdx.x, b = blockIdx.y, t = threadIdx.x;
    int c0 = chunk * 16;
    float acc = 0.f;
#pragma unroll
    for (int r = 0; r < 16; r++) {
        int c = c0 + r;
        size_t base = ((size_t)b * C_ + c) * P_;
        float a = ac[b * C_ + c];
        if (t < P_) {
            float w = img[base + t] * a;
            ldsT[r * 200 + t] = f2bf(w);
            acc += w * w;
        }
    }
    __syncthreads();
    if (t < PP_) {
        v16u vv = {};
        if (t < P_) {
#pragma unroll
            for (int r = 0; r < 16; r++) vv[r] = ldsT[r * 200 + t];
        }
        // coalesced 32B store of one transposed row-chunk (padding rows -> zeros)
        *(v16u*)(xwt + ((size_t)b * PP_ + t) * C_ + c0) = vv;
    }
    if (t < P_) sqp[((size_t)b * 128 + chunk) * P_ + t] = acc;
}

// ---- 7. rp[b,p] = 1/max(sqrt(sum sq),eps) (fixed-order reduce of partials) ----
__global__ void k_rp(const float* __restrict__ sqp, float* __restrict__ rp) {
    int i = blockIdx.x * 256 + threadIdx.x;
    if (i >= B_ * P_) return;
    int b = i / P_, p = i % P_;
    float s = 0.f;
    for (int ch = 0; ch < 128; ch++) s += sqp[((size_t)b * 128 + ch) * P_ + p];
    rp[i] = 1.f / fmaxf(sqrtf(s), EPSV);
}

// ---- 8. WMMA GEMM ws@xw: all fragments straight from global (no LDS, no
//         barriers in K-loop); fused rp scale + tanh + wsi reduction ----
__global__ void __launch_bounds__(256) k_gemm(
    const unsigned short* __restrict__ wsb, const unsigned short* __restrict__ xwt,
    const float* __restrict__ rp, const float* __restrict__ bs,
    const float* __restrict__ hsv, const float* __restrict__ wsi,
    float* __restrict__ ssp) {
    __shared__ float lds_part[8][32];
    int mt = blockIdx.x, pt = blockIdx.y, b = blockIdx.z;
    int t = threadIdx.x, w = t >> 5, l = t & 31;
    int hi = l >> 4, lo = l & 15;
    int pbase  = pt * 32;
    int m_wave = mt * 128 + w * 16;

    // A per-lane layout: lane<16 holds K {0..15}, lane>=16 holds {16..31}
    const unsigned short* aptr  = wsb + (size_t)(m_wave + lo) * C_ + 16 * hi;
    // B per-lane layout: column p fixed per lane, 16 contiguous K halves, seg = hi*16
    const unsigned short* bptr0 = xwt + ((size_t)b * PP_ + pbase + lo) * C_ + 16 * hi;
    const unsigned short* bptr1 = bptr0 + (size_t)16 * C_;

    v8f acc0 = {}; v8f acc1 = {};
#pragma unroll 4
    for (int kb = 0; kb < C_; kb += 32) {
        v8u a0 = *(const v8u*)(aptr);
        v8u a1 = *(const v8u*)(aptr + 8);
        v16u au = __builtin_shufflevector(a0, a1, 0,1,2,3,4,5,6,7,8,9,10,11,12,13,14,15);
        v16bf a = __builtin_bit_cast(v16bf, au);
        v16bf b0 = __builtin_bit_cast(v16bf, *(const v16u*)bptr0);
        v16bf b1 = __builtin_bit_cast(v16bf, *(const v16u*)bptr1);
        acc0 = __builtin_amdgcn_wmma_f32_16x16x32_bf16(false, a, false, b0, (short)0, acc0, false, false);
        acc1 = __builtin_amdgcn_wmma_f32_16x16x32_bf16(false, a, false, b1, (short)0, acc1, false, false);
        aptr += 32; bptr0 += 32; bptr1 += 32;
    }

    // epilogue: C/D layout -> lane column N = lo, row M = r + 8*hi
    float bsr[8], hsr[8], wsr[8];
#pragma unroll
    for (int r = 0; r < 8; r++) {
        int m = m_wave + r + 8 * hi;
        bsr[r] = bs[m];
        hsr[r] = hsv[b * K_ + m];
        wsr[r] = wsi[m];
    }
#pragma unroll
    for (int f = 0; f < 2; f++) {
        int p = pbase + f * 16 + lo;
        bool valid = (p < P_);
        float rpv = valid ? rp[b * P_ + p] : 0.f;
        v8f accv = f ? acc1 : acc0;
        float part = 0.f;
#pragma unroll
        for (int r = 0; r < 8; r++)
            part += wsr[r] * tanhf(accv[r] * rpv + bsr[r] + hsr[r]);
        if (!valid) part = 0.f;
        part += __shfl_xor(part, 16, 32);          // combine M halves (same column)
        if (l < 16) lds_part[w][f * 16 + l] = part;
    }
    __syncthreads();
    if (t < 32) {
        float s = 0.f;
#pragma unroll
        for (int ww = 0; ww < 8; ww++) s += lds_part[ww][t];   // fixed order: deterministic
        int p = pbase + t;
        if (p < P_) ssp[((size_t)(b * 4 + mt)) * P_ + p] = s;
    }
}

// ---- 9. combine mtile partials (+bsi), softmax over P -> asp ----
__global__ void k_smP(const float* __restrict__ ssp, const float* __restrict__ bsi,
                      float* __restrict__ asp) {
    __shared__ float red[256];
    int b = blockIdx.x, t = threadIdx.x;
    bool act = (t < P_);
    float val = 0.f, v = -1e30f;
    if (act) {
        val = bsi[0];
        for (int mt = 0; mt < 4; mt++) val += ssp[((size_t)(b * 4 + mt)) * P_ + t];
        v = val;
    }
    red[t] = v; __syncthreads();
    for (int s = 128; s > 0; s >>= 1) { if (t < s) red[t] = fmaxf(red[t], red[t + s]); __syncthreads(); }
    float mx = red[0]; __syncthreads();
    float e = act ? expf(val - mx) : 0.f;
    red[t] = e; __syncthreads();
    for (int s = 128; s > 0; s >>= 1) { if (t < s) red[t] += red[t + s]; __syncthreads(); }
    float rs = 1.f / red[0];
    if (act) asp[b * P_ + t] = e * rs;
}

// ---- 10. out = x * ac * asp  (float4: rows of 196 are divisible by 4) ----
__global__ void k_out(const float4* __restrict__ img4, const float* __restrict__ ac,
                      const float* __restrict__ asp, float4* __restrict__ out4) {
    size_t i = (size_t)blockIdx.x * 256 + threadIdx.x;          // float4 index
    size_t e = i * 4;                                           // element index
    int p = (int)(e % P_);
    size_t bc = e / P_;
    int b = (int)(bc / C_);
    float a = ac[bc];
    const float* ap = asp + b * P_ + p;
    float4 x = img4[i];
    float4 o;
    o.x = x.x * a * ap[0];
    o.y = x.y * a * ap[1];
    o.z = x.z * a * ap[2];
    o.w = x.w * a * ap[3];
    out4[i] = o;
}

extern "C" void kernel_launch(void* const* d_in, const int* in_sizes, int n_in,
                              void* d_out, int out_size, void* d_ws, size_t ws_size,
                              hipStream_t stream) {
    (void)in_sizes; (void)n_in; (void)out_size; (void)ws_size;
    const float* img = (const float*)d_in[0];
    const float* h   = (const float*)d_in[1];
    const float* wc  = (const float*)d_in[2];
    const float* bc  = (const float*)d_in[3];
    const float* whc = (const float*)d_in[4];
    const float* wci = (const float*)d_in[5];
    const float* bci = (const float*)d_in[6];
    const float* ws  = (const float*)d_in[7];
    const float* bs  = (const float*)d_in[8];
    const float* whs = (const float*)d_in[9];
    const float* wsi = (const float*)d_in[10];
    const float* bsi = (const float*)d_in[11];
    float* out = (float*)d_out;

    char* wsp = (char*)d_ws;
    float* ic   = (float*)(wsp + 0);                      // 512 KB
    float* sc   = (float*)(wsp + 524288);                 // 512 KB
    float* ac   = (float*)(wsp + 1048576);                // 512 KB
    float* hc   = (float*)(wsp + 1572864);                // 128 KB
    float* hsv  = (float*)(wsp + 1703936);                // 128 KB
    float* rp   = (float*)(wsp + 1835008);                // 49 KB
    float* asp  = (float*)(wsp + 1885184);                // 49 KB
    float* ssp  = (float*)(wsp + 1935360);                // 196 KB  [64*4*196]
    unsigned short* wsb = (unsigned short*)(wsp + 2136064);   // 2 MB   bf16 ws
    float* sqp  = (float*)(wsp + 4233216);                // 6.1 MB  [64*128*196]
    unsigned short* xwt = (unsigned short*)(wsp + 10655744);  // 58.7 MB bf16 xw^T [B][224][C]

    k_ic   <<<16384, 256, 0, stream>>>(img, ic);
    k_cvtws<<<4096, 256, 0, stream>>>(ws, wsb);
    k_hproj<<<8192, 256, 0, stream>>>(whc, whs, h, hc, hsv);
    k_sc   <<<16384, 256, 0, stream>>>(ic, hc, wc, bc, wci, bci, sc);
    k_smC  <<<64, 256, 0, stream>>>(sc, ac);
    k_xw   <<<dim3(128, 64), 256, 0, stream>>>(img, ac, xwt, sqp);
    k_rp   <<<49, 256, 0, stream>>>(sqp, rp);
    k_gemm <<<dim3(4, 7, 64), 256, 0, stream>>>(wsb, xwt, rp, bs, hsv, wsi, ssp);
    k_smP  <<<64, 256, 0, stream>>>(ssp, bsi, asp);
    k_out  <<<25088, 256, 0, stream>>>((const float4*)img, ac, asp, (float4*)out);
}
